// QNN_02_78778290143343
// MI455X (gfx1250) — matrix operand order repaired and checked by reading, not compiled, not run on hardware
//
#include <hip/hip_runtime.h>
#include <hip/hip_bf16.h>
#include <math.h>

typedef __attribute__((ext_vector_type(16))) _Float16 v16h;
typedef __attribute__((ext_vector_type(8)))  _Float16 v8h;
typedef __attribute__((ext_vector_type(8)))  float    v8f;

#define PI_F 3.14159265358979323846f

// ---------------------------------------------------------------------------
// 0) Per-sample standardization over 52 features -> f16, zero padded to 64.
// ---------------------------------------------------------------------------
__global__ __launch_bounds__(64) void qnn_normalize(const float* __restrict__ x,
                                                    _Float16* __restrict__ A0) {
  __shared__ float s1[64], s2[64];
  int j = threadIdx.x, b = blockIdx.x;
  float v = (j < 52) ? x[(size_t)b * 52 + j] : 0.f;
  s1[j] = v; s2[j] = v * v;
  __syncthreads();
  for (int s = 32; s > 0; s >>= 1) {
    if (j < s) { s1[j] += s1[j + s]; s2[j] += s2[j + s]; }
    __syncthreads();
  }
  float mu  = s1[0] * (1.f / 52.f);
  float var = s2[0] * (1.f / 52.f) - mu * mu;
  float sd  = sqrtf(fmaxf(var, 0.f));
  float inv = 1.f / (sd + 1e-8f);
  A0[(size_t)b * 64 + j] = (_Float16)((j < 52) ? (v - mu) * inv : 0.f);
}

// ---------------------------------------------------------------------------
// 1) Weight prep: f32 (K x N) -> f16 transposed+padded (Np x Kp), zeros pad.
// ---------------------------------------------------------------------------
__global__ void qnn_transpose_pad(_Float16* __restrict__ dst,
                                  const float* __restrict__ src,
                                  int K, int N, int Kp, int Np) {
  int tid = blockIdx.x * blockDim.x + threadIdx.x;
  if (tid >= Kp * Np) return;
  int n = tid / Kp, k = tid - n * Kp;
  dst[tid] = (k < K && n < N) ? (_Float16)src[(size_t)k * N + n] : (_Float16)0.f;
}

// ---------------------------------------------------------------------------
// 2) Generic WMMA GEMM: C = act(A[MxK] * Bt[NxK]^T + bias), f16 in, f32 acc.
//    8 waves per block, one 16x16 tile per wave. act: 0=none 1=elu 2=tanh*pi
// ---------------------------------------------------------------------------
__global__ __launch_bounds__(256) void qnn_wmma_gemm(
    const _Float16* __restrict__ A, const _Float16* __restrict__ Bt,
    const float* __restrict__ bias, _Float16* __restrict__ C16,
    float* __restrict__ C32, int M, int N, int K, int act, int biasN) {
  int wave = threadIdx.x >> 5;
  int lane = threadIdx.x & 31;
  int tiles_n = N >> 4;
  int tiles_total = (M >> 4) * tiles_n;
  int tile = blockIdx.x * 8 + wave;
  if (tile >= tiles_total) return;              // wave-uniform exit (EXEC stays all-1)
  int mt = tile / tiles_n, nt = tile - mt * tiles_n;
  int l16 = lane & 15, lh = lane >> 4;
  int row = (mt << 4) + l16;                    // A row this lane feeds
  int col = (nt << 4) + l16;                    // C column this lane owns

  v8f acc;
  float bv = (bias && col < biasN) ? bias[col] : 0.f;
#pragma unroll
  for (int r = 0; r < 8; ++r) acc[r] = bv;

  const _Float16* arow = A  + (size_t)row * K;
  const _Float16* brow = Bt + (size_t)col * K;

  for (int k0 = 0; k0 < K; k0 += 32) {
    // A fragment: lane-half lh holds 16 consecutive K starting at ka = k0 + lh*16
    int ka = k0 + (lh << 4);
    v8h alo = *(const v8h*)(arow + ka);
    v8h ahi = *(const v8h*)(arow + ka + 8);
    v16h af;
#pragma unroll
    for (int e = 0; e < 8; ++e) { af[e] = alo[e]; af[e + 8] = ahi[e]; }
    // B fragment: lane-half lh holds 16 consecutive K starting at k0 + lh*16
    v16h bf = *(const v16h*)(brow + k0 + (lh << 4));
    acc = __builtin_amdgcn_wmma_f32_16x16x32_f16(false, af, false, bf,
                                                 (short)0, acc, false, false);
  }

#pragma unroll
  for (int r = 0; r < 8; ++r) {
    int m = (mt << 4) + r + (lh << 3);
    float v = acc[r];
    if (act == 1)      v = v > 0.f ? v : expm1f(v);
    else if (act == 2) v = tanhf(v) * PI_F;
    if (C16) C16[(size_t)m * N + col] = (_Float16)v;
    if (C32) C32[(size_t)m * N + col] = v;
  }
}

// ---------------------------------------------------------------------------
// 3) 8-qubit statevector simulator. One block (128 threads) per sample.
//    Wire i <-> bit position (7-i) of the flat index. psi lives in LDS.
// ---------------------------------------------------------------------------
__device__ __forceinline__ float2 cxmul(float2 a, float2 b) {
  return make_float2(a.x * b.x - a.y * b.y, a.x * b.y + a.y * b.x);
}

__global__ __launch_bounds__(128) void qnn_qsim(const float* __restrict__ angles,
                                                const float* __restrict__ qw,
                                                _Float16* __restrict__ Aq) {
  __shared__ float2 psi[256];
  __shared__ float sx[128], sy[128], sz[128];
  __shared__ float qout[24];
  int t = threadIdx.x;
  int b = blockIdx.x;
  const float* ang = angles + (size_t)b * 64;

  psi[t]       = make_float2(t == 0 ? 1.f : 0.f, 0.f);
  psi[t + 128] = make_float2(0.f, 0.f);

#pragma unroll 1
  for (int l = 0; l < 7; ++l) {
    // --- fused RY(chunk) followed by Rot(phi,theta,omega), per qubit ---
#pragma unroll 1
    for (int i = 0; i < 8; ++i) {
      int idx = l * 8 + i;
      float th_ry = (idx < 52) ? ang[idx] : 0.f;
      float phi = qw[idx * 3 + 0], th = qw[idx * 3 + 1], om = qw[idx * 3 + 2];
      float cr = cosf(0.5f * th_ry), sr = sinf(0.5f * th_ry);
      float ct = cosf(0.5f * th),    st = sinf(0.5f * th);
      float ap = 0.5f * (phi + om),  am = 0.5f * (phi - om);
      float cap = cosf(ap), sap = sinf(ap), cam = cosf(am), sam = sinf(am);
      // Rot = RZ(om) RY(th) RZ(phi)
      float2 r00 = make_float2( ct * cap, -ct * sap);
      float2 r01 = make_float2(-st * cam, -st * sam);
      float2 r10 = make_float2( st * cam, -st * sam);
      float2 r11 = make_float2( ct * cap,  ct * sap);
      // U = Rot * RY(th_ry)
      float2 u00 = make_float2( r00.x * cr + r01.x * sr,  r00.y * cr + r01.y * sr);
      float2 u01 = make_float2(-r00.x * sr + r01.x * cr, -r00.y * sr + r01.y * cr);
      float2 u10 = make_float2( r10.x * cr + r11.x * sr,  r10.y * cr + r11.y * sr);
      float2 u11 = make_float2(-r10.x * sr + r11.x * cr, -r10.y * sr + r11.y * cr);

      int p = 7 - i, pb = 1 << p;
      int i0 = ((t >> p) << (p + 1)) | (t & (pb - 1));
      int i1 = i0 | pb;
      __syncthreads();                      // previous gate fully written
      float2 a0 = psi[i0], a1 = psi[i1];    // pair owned exclusively by this thread
      float2 n0 = make_float2(u00.x * a0.x - u00.y * a0.y + u01.x * a1.x - u01.y * a1.y,
                              u00.x * a0.y + u00.y * a0.x + u01.x * a1.y + u01.y * a1.x);
      float2 n1 = make_float2(u10.x * a0.x - u10.y * a0.y + u11.x * a1.x - u11.y * a1.y,
                              u10.x * a0.y + u10.y * a0.x + u11.x * a1.y + u11.y * a1.x);
      psi[i0] = n0; psi[i1] = n1;
    }

    // --- CNOT entanglers ---
    auto do_cnot = [&](int c, int tq) {
      __syncthreads();
      if (t < 64) {
        int pc = 7 - c, pt = 7 - tq;
        int p1 = pc > pt ? pc : pt, p0 = pc > pt ? pt : pc;
        int low  = t & ((1 << p0) - 1);
        int mid  = (t >> p0) & ((1 << (p1 - 1 - p0)) - 1);
        int high = t >> (p1 - 1);
        int idx = (high << (p1 + 1)) | (mid << (p0 + 1)) | low;
        idx |= (1 << pc);                   // control = 1, target = 0
        int part = idx | (1 << pt);
        float2 tmp = psi[idx]; psi[idx] = psi[part]; psi[part] = tmp;
      }
    };
    if ((l & 1) == 0) {
      for (int i = 0; i < 7; ++i) do_cnot(i, i + 1);
      do_cnot(7, 0);
    } else {
      for (int i = 0; i < 7; i += 2) do_cnot(i, i + 1);
    }

    // --- all 8 RZ(0.5*chunk) fused into one diagonal phase pass ---
    __syncthreads();
#pragma unroll 1
    for (int rep = 0; rep < 2; ++rep) {
      int idx = t + (rep << 7);
      float ph = 0.f;
      for (int i = 0; i < 8; ++i) {
        int ii = l * 8 + i;
        float a = 0.25f * ((ii < 52) ? ang[ii] : 0.f);   // RZ(0.5*c) -> half angle 0.25*c
        ph += ((idx >> (7 - i)) & 1) ? a : -a;
      }
      psi[idx] = cxmul(psi[idx], make_float2(cosf(ph), sinf(ph)));
    }
  }

  // --- expectations <X_i>, <Y_i>, <Z_i> ---
  __syncthreads();
#pragma unroll 1
  for (int i = 0; i < 8; ++i) {
    int p = 7 - i, pb = 1 << p;
    int i0 = ((t >> p) << (p + 1)) | (t & (pb - 1));
    int i1 = i0 | pb;
    float2 a0 = psi[i0], a1 = psi[i1];
    sx[t] = a0.x * a1.x + a0.y * a1.y;                                   // Re(conj(a0)a1)
    sy[t] = a0.x * a1.y - a0.y * a1.x;                                   // Im(conj(a0)a1)
    sz[t] = (a0.x * a0.x + a0.y * a0.y) - (a1.x * a1.x + a1.y * a1.y);
    __syncthreads();
    for (int s = 64; s > 0; s >>= 1) {
      if (t < s) { sx[t] += sx[t + s]; sy[t] += sy[t + s]; sz[t] += sz[t + s]; }
      __syncthreads();
    }
    if (t == 0) {
      qout[3 * i + 0] = 2.f * sx[0];
      qout[3 * i + 1] = 2.f * sy[0];
      qout[3 * i + 2] = sz[0];
    }
    __syncthreads();
  }
  if (t < 32) Aq[(size_t)b * 32 + t] = (_Float16)(t < 24 ? qout[t] : 0.f);
}

// ---------------------------------------------------------------------------
// 4) Final heads: (F,B,64) f16 @ hw2[f](64x2) + hb2 -> mus, softplus(sig)
// ---------------------------------------------------------------------------
__global__ void qnn_head_final(const _Float16* __restrict__ fo,
                               const float* __restrict__ hw2,
                               const float* __restrict__ hb2,
                               float* __restrict__ out, int B) {
  int tid = blockIdx.x * blockDim.x + threadIdx.x;
  if (tid >= B * 6) return;
  int b = tid / 6, f = tid - b * 6;
  const _Float16* fr = fo + ((size_t)f * B + b) * 64;
  const float* w = hw2 + (size_t)f * 64 * 2;
  float a0 = hb2[f * 2 + 0], a1 = hb2[f * 2 + 1];
#pragma unroll 8
  for (int h = 0; h < 64; ++h) {
    float v = (float)fr[h];
    a0 += v * w[h * 2 + 0];
    a1 += v * w[h * 2 + 1];
  }
  out[(size_t)b * 6 + f] = a0;
  float sp = (a1 > 0.f) ? a1 + log1pf(expf(-a1)) : log1pf(expf(a1));
  out[(size_t)B * 6 + b * 6 + f] = sp;
}

// ---------------------------------------------------------------------------
// Host-side pipeline
// ---------------------------------------------------------------------------
static inline void launch_gemm(const _Float16* A, const _Float16* Bt,
                               const float* bias, _Float16* C16, float* C32,
                               int M, int N, int K, int act, int biasN,
                               hipStream_t stream) {
  int tiles = (M / 16) * (N / 16);
  int blocks = (tiles + 7) / 8;
  qnn_wmma_gemm<<<blocks, 256, 0, stream>>>(A, Bt, bias, C16, C32, M, N, K, act, biasN);
}

extern "C" void kernel_launch(void* const* d_in, const int* in_sizes, int n_in,
                              void* d_out, int out_size, void* d_ws, size_t ws_size,
                              hipStream_t stream) {
  (void)in_sizes; (void)n_in; (void)out_size; (void)ws_size;
  const float* x     = (const float*)d_in[0];
  const float* enc_w = (const float*)d_in[1];
  const float* enc_b = (const float*)d_in[2];
  const float* qw    = (const float*)d_in[3];
  const float* w0    = (const float*)d_in[4];
  const float* b0    = (const float*)d_in[5];
  const float* w1    = (const float*)d_in[6];
  const float* b1    = (const float*)d_in[7];
  const float* w2    = (const float*)d_in[8];
  const float* b2    = (const float*)d_in[9];
  const float* hw1   = (const float*)d_in[10];
  const float* hb1   = (const float*)d_in[11];
  const float* hw2   = (const float*)d_in[12];
  const float* hb2   = (const float*)d_in[13];
  float* out = (float*)d_out;
  const int B = 4096;

  char* base = (char*)d_ws;
  size_t off = 0;
  auto alloc = [&](size_t bytes) -> void* {
    void* p = base + off;
    off += (bytes + 255) & ~(size_t)255;
    return p;
  };
  _Float16* A0     = (_Float16*)alloc((size_t)B * 64 * 2);   // normalized inputs, pad 52->64
  _Float16* enc_wt = (_Float16*)alloc(64 * 64 * 2);
  float*    angles = (float*)   alloc((size_t)B * 64 * 4);
  _Float16* Aq     = (_Float16*)alloc((size_t)B * 32 * 2);   // expectations, pad 24->32
  _Float16* w0t    = (_Float16*)alloc(256 * 32 * 2);
  _Float16* h0     = (_Float16*)alloc((size_t)B * 256 * 2);
  _Float16* w1t    = (_Float16*)alloc(256 * 256 * 2);
  _Float16* h1     = (_Float16*)alloc((size_t)B * 256 * 2);
  _Float16* w2t    = (_Float16*)alloc(128 * 256 * 2);
  _Float16* h2     = (_Float16*)alloc((size_t)B * 128 * 2);
  _Float16* hw1t   = (_Float16*)alloc((size_t)6 * 64 * 128 * 2);
  _Float16* fo     = (_Float16*)alloc((size_t)6 * B * 64 * 2);

  // Stage 0: normalization + weight prep
  qnn_normalize<<<B, 64, 0, stream>>>(x, A0);
  qnn_transpose_pad<<<(64 * 64 + 255) / 256, 256, 0, stream>>>(enc_wt, enc_w, 52, 52, 64, 64);
  qnn_transpose_pad<<<(256 * 32 + 255) / 256, 256, 0, stream>>>(w0t, w0, 24, 256, 32, 256);
  qnn_transpose_pad<<<(256 * 256 + 255) / 256, 256, 0, stream>>>(w1t, w1, 256, 256, 256, 256);
  qnn_transpose_pad<<<(128 * 256 + 255) / 256, 256, 0, stream>>>(w2t, w2, 256, 128, 256, 128);
  for (int f = 0; f < 6; ++f)
    qnn_transpose_pad<<<(64 * 128 + 255) / 256, 256, 0, stream>>>(
        hw1t + (size_t)f * 64 * 128, hw1 + (size_t)f * 128 * 64, 128, 64, 128, 64);

  // Stage 1: encoder GEMM + tanh*pi -> angles (f32)
  launch_gemm(A0, enc_wt, enc_b, nullptr, angles, B, 64, 64, /*act=*/2, /*biasN=*/52, stream);

  // Stage 2: quantum circuit -> expectations (f16, padded to 32)
  qnn_qsim<<<B, 128, 0, stream>>>(angles, qw, Aq);

  // Stage 3: MLP with fused ELU (all WMMA)
  launch_gemm(Aq, w0t, b0, h0, nullptr, B, 256, 32, 1, 256, stream);
  launch_gemm(h0, w1t, b1, h1, nullptr, B, 256, 256, 1, 256, stream);
  launch_gemm(h1, w2t, b2, h2, nullptr, B, 128, 256, 1, 128, stream);

  // Stage 4: per-feature head hidden layers (WMMA, 6 slices)
  for (int f = 0; f < 6; ++f)
    launch_gemm(h2, hw1t + (size_t)f * 64 * 128, hb1 + (size_t)f * 64,
                fo + (size_t)f * B * 64, nullptr, B, 64, 128, 1, 64, stream);

  // Stage 5: final 64->2 heads, mus + softplus
  qnn_head_final<<<(B * 6 + 255) / 256, 256, 0, stream>>>(fo, hw2, hb2, out, B);
}
